// SimpleGRU_65644280152709
// MI455X (gfx1250) — matrix operand order repaired and checked by reading, not compiled, not run on hardware
//
#include <hip/hip_runtime.h>
#include <hip/hip_bf16.h>
#include <math.h>

typedef __attribute__((ext_vector_type(16))) __bf16 v16bf;
typedef __attribute__((ext_vector_type(8)))  __bf16 v8bf;
typedef __attribute__((ext_vector_type(8)))  float  v8f;

#define NWAVES 8
#define BTILE  32
#define NTHREADS 256

// LDS layout (dynamic shared):
//   s_in : bf16 [32][672]   (3 timesteps, K padded to 224)  43008 B @ 0
//   s_mx : f32  [32][384]                                   49152 B @ 43008
//   s_mi : f32  [32][384]                                   49152 B @ 92160
//   s_h  : bf16 [32][128]                                    8192 B @ 141312
#define SMEM_BYTES 149504

// C = A(LDS, [32 x *], row-major bf16) x B(global, transposed bf16 [N x Kp]) -> S (LDS f32, stride 384)
// One wave computes 16x16 tiles; 2 M-tiles x NT N-tiles split across 8 waves.
template <int KC>
__device__ __forceinline__ void gemm_bf16(const __bf16* __restrict__ As, int aStride, int aCol0,
                                          const __bf16* __restrict__ W, int wStride,
                                          float* __restrict__ S, int NT, int wave, int lane) {
  const int half = lane >> 4;   // 0: lanes 0-15, 1: lanes 16-31
  const int l16  = lane & 15;
  const int tiles = 2 * NT;
  for (int tile = wave; tile < tiles; tile += NWAVES) {
    const int mt = tile / NT;
    const int nt = tile - mt * NT;
    // A fragment base: lane half 0 holds K {0..7,16..23}, half 1 holds K {8..15,24..31}
    const __bf16* a0 = As + (mt * 16 + l16) * aStride + aCol0 + (half ? 8 : 0);
    // B fragment base (transposed weights): lane half 0 holds K {0..7,16..23}, half 1 holds K {8..15,24..31} of column n
    const __bf16* b0 = W + (nt * 16 + l16) * wStride + half * 8;
    __builtin_prefetch(b0, 0, 1);
    v8f acc = {};
#pragma unroll
    for (int kc = 0; kc < KC; ++kc) {
      v16bf a, b;
      *((v8bf*)&a + 0) = *(const v8bf*)(a0 + kc * 32);        // ds_load_b128
      *((v8bf*)&a + 1) = *(const v8bf*)(a0 + kc * 32 + 16);
      *((v8bf*)&b + 0) = *(const v8bf*)(b0 + kc * 32);        // global_load_b128
      *((v8bf*)&b + 1) = *(const v8bf*)(b0 + kc * 32 + 16);
      acc = __builtin_amdgcn_wmma_f32_16x16x32_bf16(false, a, false, b, (short)0, acc,
                                                    false, false);
    }
    // C/D layout: VGPR v, lanes 0-15 -> (M=v, N=lane); lanes 16-31 -> (M=v+8, N=lane-16)
    float* d = S + (mt * 16 + 8 * half) * 384 + nt * 16 + l16;
#pragma unroll
    for (int v = 0; v < 8; ++v) d[v * 384] = acc[v];
  }
}

// One GRU layer over T=3 for a 32-row batch tile, fully LDS-resident.
template <int U, int KPIN, int KCIN, int KPNEXT>
__device__ __forceinline__ void run_layer(__bf16* s_in, float* s_mx, float* s_mi, __bf16* s_h,
                                          const __bf16* wt, const __bf16* wtr,
                                          const float* bias, int tid, int wave, int lane) {
  constexpr int N3  = 3 * U;
  constexpr int NT  = N3 / 16;
  constexpr int KCH = U / 32;
  const float* bin  = bias;
  const float* brec = bias + N3;

  for (int idx = tid; idx < BTILE * U; idx += NTHREADS) s_h[idx] = (__bf16)0.0f;
  __syncthreads();

  for (int t = 0; t < 3; ++t) {
    gemm_bf16<KCIN>(s_in, 672, t * KPIN, wt, KPIN, s_mx, NT, wave, lane);  // MX = x_t @ K
    gemm_bf16<KCH>(s_h, U, 0, wtr, U, s_mi, NT, wave, lane);               // MI = h   @ RK
    __syncthreads();
    for (int f = tid; f < BTILE * U; f += NTHREADS) {
      const int m = f / U;
      const int u = f - m * U;
      const float* mxr = s_mx + m * 384;
      const float* mir = s_mi + m * 384;
      const float xz = mxr[u]         + bin[u];
      const float iz = mir[u]         + brec[u];
      const float xr = mxr[U + u]     + bin[U + u];
      const float ir = mir[U + u]     + brec[U + u];
      const float xh = mxr[2 * U + u] + bin[2 * U + u];
      const float ih = mir[2 * U + u] + brec[2 * U + u];
      const float z  = 1.0f / (1.0f + __expf(-(xz + iz)));
      const float r  = 1.0f / (1.0f + __expf(-(xr + ir)));
      const float hh = tanhf(xh + r * ih);            // reset_after=True
      const float h0 = (float)s_h[f];
      const float hn = z * h0 + (1.0f - z) * hh;
      s_h[f] = (__bf16)hn;
      if (KPNEXT > 0) s_in[m * 672 + t * KPNEXT + u] = (__bf16)hn;  // in-place: safe, KPNEXT<KPIN
    }
    __syncthreads();
  }
}

__global__ __launch_bounds__(NTHREADS, 1) void gru_stack_kernel(
    const float* __restrict__ x,
    const __bf16* wt1, const __bf16* wtr1, const float* b1,
    const __bf16* wt2, const __bf16* wtr2, const float* b2,
    const __bf16* wt3, const __bf16* wtr3, const float* b3,
    const float* Wd, const float* bd, float* __restrict__ out) {
  extern __shared__ char smem[];
  __bf16* s_in = (__bf16*)(smem);
  float*  s_mx = (float*)(smem + 43008);
  float*  s_mi = (float*)(smem + 92160);
  __bf16* s_h  = (__bf16*)(smem + 141312);

  const int tid  = threadIdx.x;
  const int wave = tid >> 5;
  const int lane = tid & 31;
  const int row0 = blockIdx.x * BTILE;

  // Stage x tile: f32 -> bf16, K padded 223 -> 224 (zero pad column)
  for (int idx = tid; idx < BTILE * 672; idx += NTHREADS) {
    const int m = idx / 672;
    const int c = idx - m * 672;
    const int t = c / 224;
    const int k = c - t * 224;
    float v = 0.0f;
    if (k < 223) v = x[((size_t)(row0 + m) * 3 + t) * 223 + k];
    s_in[m * 672 + c] = (__bf16)v;
  }

  run_layer<128, 224, 7, 128>(s_in, s_mx, s_mi, s_h, wt1, wtr1, b1, tid, wave, lane);
  run_layer<64, 128, 4, 64>(s_in, s_mx, s_mi, s_h, wt2, wtr2, b2, tid, wave, lane);
  run_layer<32, 64, 2, 0>(s_in, s_mx, s_mi, s_h, wt3, wtr3, b3, tid, wave, lane);

  // Dense(17) + softmax, one row per lane (tiny: 544 MACs/row)
  if (tid < BTILE) {
    const int m = tid;
    float logits[17];
    float mmax = -3.0e38f;
#pragma unroll
    for (int c = 0; c < 17; ++c) {
      float s = bd[c];
#pragma unroll
      for (int u = 0; u < 32; ++u) s += (float)s_h[m * 32 + u] * Wd[u * 17 + c];
      logits[c] = s;
      mmax = fmaxf(mmax, s);
    }
    float sum = 0.0f;
#pragma unroll
    for (int c = 0; c < 17; ++c) { logits[c] = __expf(logits[c] - mmax); sum += logits[c]; }
    const float inv = 1.0f / sum;
#pragma unroll
    for (int c = 0; c < 17; ++c) out[(size_t)(row0 + m) * 17 + c] = logits[c] * inv;
  }
}

// Weight prep: f32 [K x N] row-major -> bf16 transposed [N x Kp], K zero-padded to Kp.
__global__ __launch_bounds__(256) void cvt_tr_kernel(const float* __restrict__ src,
                                                     __bf16* __restrict__ dst,
                                                     int K, int N3, int Kp) {
  const int idx = blockIdx.x * 256 + threadIdx.x;
  if (idx >= N3 * Kp) return;
  const int n = idx / Kp;
  const int k = idx - n * Kp;
  const float v = (k < K) ? src[k * N3 + n] : 0.0f;
  dst[idx] = (__bf16)v;
}

extern "C" void kernel_launch(void* const* d_in, const int* in_sizes, int n_in,
                              void* d_out, int out_size, void* d_ws, size_t ws_size,
                              hipStream_t stream) {
  (void)in_sizes; (void)n_in; (void)out_size; (void)ws_size;
  const float* x   = (const float*)d_in[0];
  const float* k1  = (const float*)d_in[1];
  const float* rk1 = (const float*)d_in[2];
  const float* b1  = (const float*)d_in[3];
  const float* k2  = (const float*)d_in[4];
  const float* rk2 = (const float*)d_in[5];
  const float* b2  = (const float*)d_in[6];
  const float* k3  = (const float*)d_in[7];
  const float* rk3 = (const float*)d_in[8];
  const float* b3  = (const float*)d_in[9];
  const float* Wd  = (const float*)d_in[10];
  const float* bd  = (const float*)d_in[11];

  char* ws = (char*)d_ws;  // 362,496 bytes used, all offsets 16B aligned
  __bf16* wt1  = (__bf16*)(ws + 0);        // 384 x 224
  __bf16* wtr1 = (__bf16*)(ws + 172032);   // 384 x 128
  __bf16* wt2  = (__bf16*)(ws + 270336);   // 192 x 128
  __bf16* wtr2 = (__bf16*)(ws + 319488);   // 192 x 64
  __bf16* wt3  = (__bf16*)(ws + 344064);   //  96 x 64
  __bf16* wtr3 = (__bf16*)(ws + 356352);   //  96 x 32

  auto cvt = [&](const float* s, __bf16* d, int K, int N3, int Kp) {
    const int total = N3 * Kp;
    cvt_tr_kernel<<<(total + 255) / 256, 256, 0, stream>>>(s, d, K, N3, Kp);
  };
  cvt(k1, wt1, 223, 384, 224);
  cvt(rk1, wtr1, 128, 384, 128);
  cvt(k2, wt2, 128, 192, 128);
  cvt(rk2, wtr2, 64, 192, 64);
  cvt(k3, wt3, 64, 96, 64);
  cvt(rk3, wtr3, 32, 96, 32);

  const int B = 131072;
  gru_stack_kernel<<<B / BTILE, NTHREADS, SMEM_BYTES, stream>>>(
      x, wt1, wtr1, b1, wt2, wtr2, b2, wt3, wtr3, b3, Wd, bd, (float*)d_out);
}
